// FeedForward_37125697307132
// MI455X (gfx1250) — 8-bit matrix operand order repaired and checked by reading, compiled, not run on hardware
//
#include <hip/hip_runtime.h>


#define T_TOK 8192
#define E_DIM 2048
#define F_DIM 8192
#define EPSQ  1e-5f
#define W_CNT 16777216.0f

typedef __attribute__((ext_vector_type(8))) int v8i;

__device__ inline unsigned lds_off(const void* p) {
  return (unsigned)(size_t)p;
}

#define ASYNC_B128(ldsoff_, gptr_)                                        \
  asm volatile("global_load_async_to_lds_b128 %0, %1, off"                \
               :: "v"(ldsoff_),                                           \
                  "v"((unsigned long long)(size_t)(gptr_))                \
               : "memory")

#define WAIT_ASYNC(n_) asm volatile("s_wait_asynccnt " #n_ ::: "memory")


__device__ inline unsigned short f32_to_bf16(float f) {
  unsigned int u = __float_as_uint(f);
  unsigned int r = u + 0x7FFFu + ((u >> 16) & 1u);
  return (unsigned short)(r >> 16);
}

__device__ inline float bf16_lo(unsigned int w) { return __uint_as_float(w << 16); }
__device__ inline float bf16_hi(unsigned int w) { return __uint_as_float(w & 0xFFFF0000u); }

__device__ inline float blk_reduce_sum(float v, float* buf) {
  #pragma unroll
  for (int o = 16; o > 0; o >>= 1) v += __shfl_down(v, o, 32);
  const int lane = threadIdx.x & 31, w = threadIdx.x >> 5;
  __syncthreads();
  if (lane == 0) buf[w] = v;
  __syncthreads();
  float r = buf[0];
  #pragma unroll
  for (int i = 1; i < 8; i++) r += buf[i];
  return r;
}

__device__ inline float blk_reduce_max(float v, float* buf) {
  #pragma unroll
  for (int o = 16; o > 0; o >>= 1) v = fmaxf(v, __shfl_down(v, o, 32));
  const int lane = threadIdx.x & 31, w = threadIdx.x >> 5;
  __syncthreads();
  if (lane == 0) buf[w] = v;
  __syncthreads();
  float r = buf[0];
  #pragma unroll
  for (int i = 1; i < 8; i++) r = fmaxf(r, buf[i]);
  return r;
}


__global__ void zero_kernel(float* p) {
  if (threadIdx.x < 64) p[threadIdx.x] = 0.0f;
}

__global__ __launch_bounds__(256)
void absum_kernel(const float* __restrict__ w, float* __restrict__ out, long n4) {
  __shared__ float red[8];
  float s = 0.0f;
  const float4* w4 = (const float4*)w;
  for (long i = (long)blockIdx.x * blockDim.x + threadIdx.x; i < n4;
       i += (long)gridDim.x * blockDim.x) {
    float4 f = w4[i];
    s += fabsf(f.x) + fabsf(f.y) + fabsf(f.z) + fabsf(f.w);
  }
  float tot = blk_reduce_sum(s, red);
  if (threadIdx.x == 0) atomicAdd(out, tot);
}

__global__ __launch_bounds__(256)
void wquant_kernel(const float* __restrict__ w, signed char* __restrict__ wq,
                   const float* __restrict__ sumptr) {
  const float mean = fmaxf(sumptr[0] * (1.0f / W_CNT), EPSQ);
  const float scale = 1.0f / mean;
  const long i = ((long)blockIdx.x * 256 + threadIdx.x) * 4;
  float4 f = *(const float4*)(w + i);
  char4 q;
  q.x = (signed char)fminf(fmaxf(rintf(f.x * scale), -1.0f), 1.0f);
  q.y = (signed char)fminf(fmaxf(rintf(f.y * scale), -1.0f), 1.0f);
  q.z = (signed char)fminf(fmaxf(rintf(f.z * scale), -1.0f), 1.0f);
  q.w = (signed char)fminf(fmaxf(rintf(f.w * scale), -1.0f), 1.0f);
  *(char4*)(wq + i) = q;
}

__global__ __launch_bounds__(256)
void act_quant_x_kernel(const float* __restrict__ x, signed char* __restrict__ xq,
                        float* __restrict__ xscale) {
  __shared__ float red[8];
  const int t = blockIdx.x;
  const int tid = threadIdx.x;
  const float* row = x + (long)t * E_DIM;
  const float4* rp = (const float4*)(row + tid * 8);
  float4 f0 = rp[0], f1 = rp[1];
  float vals[8] = {f0.x, f0.y, f0.z, f0.w, f1.x, f1.y, f1.z, f1.w};
  float ss = 0.0f;
  #pragma unroll
  for (int i = 0; i < 8; i++) ss += vals[i] * vals[i];
  const float tot = blk_reduce_sum(ss, red);
  const float inv = rsqrtf(tot * (1.0f / E_DIM) + 1e-6f);
  float am = 0.0f;
  #pragma unroll
  for (int i = 0; i < 8; i++) { vals[i] *= inv; am = fmaxf(am, fabsf(vals[i])); }
  const float amax = blk_reduce_max(am, red);
  const float dq = fmaxf(amax, EPSQ) * (1.0f / 127.0f);
  const float qs = 1.0f / dq;
  alignas(8) signed char qb[8];
  #pragma unroll
  for (int i = 0; i < 8; i++) {
    float q = fminf(fmaxf(rintf(vals[i] * qs), -128.0f), 127.0f);
    qb[i] = (signed char)q;
  }
  *(int2*)(xq + (long)t * E_DIM + tid * 8) = *(const int2*)qb;
  if (tid == 0) xscale[t] = dq;
}

__global__ __launch_bounds__(256)
void act_quant_h_kernel(const unsigned short* __restrict__ h,
                        signed char* __restrict__ hq, float* __restrict__ hscale) {
  __shared__ float red[8];
  const int t = blockIdx.x;
  const int tid = threadIdx.x;
  const int4* rp = (const int4*)(h + (long)t * F_DIM + tid * 32);
  float vals[32];
  float ss = 0.0f;
  #pragma unroll
  for (int j = 0; j < 4; j++) {
    int4 blk = rp[j];
    unsigned int wds[4] = {(unsigned)blk.x, (unsigned)blk.y, (unsigned)blk.z, (unsigned)blk.w};
    #pragma unroll
    for (int q = 0; q < 4; q++) {
      float lo = bf16_lo(wds[q]), hi = bf16_hi(wds[q]);
      vals[j * 8 + q * 2] = lo;
      vals[j * 8 + q * 2 + 1] = hi;
      ss += lo * lo + hi * hi;
    }
  }
  const float tot = blk_reduce_sum(ss, red);
  const float inv = rsqrtf(tot * (1.0f / F_DIM) + 1e-6f);
  float am = 0.0f;
  #pragma unroll
  for (int i = 0; i < 32; i++) { vals[i] *= inv; am = fmaxf(am, fabsf(vals[i])); }
  const float amax = blk_reduce_max(am, red);
  const float dq = fmaxf(amax, EPSQ) * (1.0f / 127.0f);
  const float qs = 1.0f / dq;
  alignas(16) signed char qb[32];
  #pragma unroll
  for (int i = 0; i < 32; i++) {
    float q = fminf(fmaxf(rintf(vals[i] * qs), -128.0f), 127.0f);
    qb[i] = (signed char)q;
  }
  int4* op = (int4*)(hq + (long)t * F_DIM + tid * 32);
  op[0] = *(const int4*)&qb[0];
  op[1] = *(const int4*)&qb[16];
  if (tid == 0) hscale[t] = dq;
}


__global__ __launch_bounds__(256)
void gemm12_silu_kernel(const signed char* __restrict__ xq,
                        const signed char* __restrict__ wq1,
                        const signed char* __restrict__ wq2,
                        const float* __restrict__ xscale,
                        const float* __restrict__ scal,
                        unsigned short* __restrict__ h) {
  __shared__ int sA[2][128 * 20];
  __shared__ int sB1[2][64 * 20];
  __shared__ int sB2[2][64 * 20];

  const int tid = threadIdx.x;
  const int lane = tid & 31;
  const int wave = tid >> 5;
  const int wmq = wave >> 1;
  const int wn = wave & 1;
  const int half = lane >> 4;
  const int l16 = lane & 15;

  const int tokenBase = blockIdx.y << 7;
  const int ffBase = blockIdx.x << 6;

  const int arow = tid >> 1;
  const int acol = (tid & 1) << 5;
  const signed char* ga = xq + (long)(tokenBase + arow) * E_DIM + acol;
  const int sIdxA = arow * 20 + ((tid & 1) << 3);

  const int brow = tid >> 2;
  const int bcol = (tid & 3) << 4;
  const signed char* gb1 = wq1 + (long)(ffBase + brow) * E_DIM + bcol;
  const signed char* gb2 = wq2 + (long)(ffBase + brow) * E_DIM + bcol;
  const int sIdxB = brow * 20 + ((tid & 3) << 2);

  const unsigned lA[2] = {lds_off(&sA[0][sIdxA]), lds_off(&sA[1][sIdxA])};
  const unsigned lB1[2] = {lds_off(&sB1[0][sIdxB]), lds_off(&sB1[1][sIdxB])};
  const unsigned lB2[2] = {lds_off(&sB2[0][sIdxB]), lds_off(&sB2[1][sIdxB])};

  const int aFrag0 = (wmq * 32 + l16) * 20 + half * 4;
  const int aFrag1 = aFrag0 + 16 * 20;
  const int bFrag0 = (wn * 32 + l16) * 20 + half * 4;
  const int bFrag1 = bFrag0 + 16 * 20;

  v8i z = {0, 0, 0, 0, 0, 0, 0, 0};
  v8i acc100 = z, acc101 = z, acc110 = z, acc111 = z;
  v8i acc200 = z, acc201 = z, acc210 = z, acc211 = z;

  ASYNC_B128(lA[0], ga);
  ASYNC_B128(lA[0] + 16u, ga + 16);
  ASYNC_B128(lB1[0], gb1);
  ASYNC_B128(lB2[0], gb2);

  const int NK = E_DIM / 64;
  for (int i = 0; i < NK; ++i) {
    const int cur = i & 1;
    if (i + 1 < NK) {
      const int nxt = cur ^ 1;
      const int kb = (i + 1) * 64;
      ASYNC_B128(lA[nxt], ga + kb);
      ASYNC_B128(lA[nxt] + 16u, ga + kb + 16);
      ASYNC_B128(lB1[nxt], gb1 + kb);
      ASYNC_B128(lB2[nxt], gb2 + kb);
      WAIT_ASYNC(0x4);
    } else {
      WAIT_ASYNC(0x0);
    }
    __syncthreads();

    const int* As = sA[cur];
    const int* Bs1 = sB1[cur];
    const int* Bs2 = sB2[cur];

    int2 q0 = *(const int2*)&As[aFrag0 + 0];
    int2 q1 = *(const int2*)&As[aFrag0 + 2];
    int2 q2 = *(const int2*)&As[aFrag0 + 8];
    int2 q3 = *(const int2*)&As[aFrag0 + 10];
    v8i a0 = {q0.x, q0.y, q1.x, q1.y, q2.x, q2.y, q3.x, q3.y};
    q0 = *(const int2*)&As[aFrag1 + 0];
    q1 = *(const int2*)&As[aFrag1 + 2];
    q2 = *(const int2*)&As[aFrag1 + 8];
    q3 = *(const int2*)&As[aFrag1 + 10];
    v8i a1 = {q0.x, q0.y, q1.x, q1.y, q2.x, q2.y, q3.x, q3.y};

    int4 p0 = *(const int4*)&Bs1[bFrag0];
    int4 p1 = *(const int4*)&Bs1[bFrag0 + 8];
    int4 p2 = *(const int4*)&Bs1[bFrag1];
    int4 p3 = *(const int4*)&Bs1[bFrag1 + 8];
    int4 p4 = *(const int4*)&Bs2[bFrag0];
    int4 p5 = *(const int4*)&Bs2[bFrag0 + 8];
    int4 p6 = *(const int4*)&Bs2[bFrag1];
    int4 p7 = *(const int4*)&Bs2[bFrag1 + 8];
    v8i b10 = {p0.x, p0.y, p0.z, p0.w, p1.x, p1.y, p1.z, p1.w};
    v8i b11 = {p2.x, p2.y, p2.z, p2.w, p3.x, p3.y, p3.z, p3.w};
    v8i b20 = {p4.x, p4.y, p4.z, p4.w, p5.x, p5.y, p5.z, p5.w};
    v8i b21 = {p6.x, p6.y, p6.z, p6.w, p7.x, p7.y, p7.z, p7.w};

    acc100 = __builtin_amdgcn_wmma_i32_16x16x64_iu8(true, a0, true, b10, acc100, false, false);
    acc101 = __builtin_amdgcn_wmma_i32_16x16x64_iu8(true, a0, true, b11, acc101, false, false);
    acc200 = __builtin_amdgcn_wmma_i32_16x16x64_iu8(true, a0, true, b20, acc200, false, false);
    acc201 = __builtin_amdgcn_wmma_i32_16x16x64_iu8(true, a0, true, b21, acc201, false, false);
    acc110 = __builtin_amdgcn_wmma_i32_16x16x64_iu8(true, a1, true, b10, acc110, false, false);
    acc111 = __builtin_amdgcn_wmma_i32_16x16x64_iu8(true, a1, true, b11, acc111, false, false);
    acc210 = __builtin_amdgcn_wmma_i32_16x16x64_iu8(true, a1, true, b20, acc210, false, false);
    acc211 = __builtin_amdgcn_wmma_i32_16x16x64_iu8(true, a1, true, b21, acc211, false, false);

    __syncthreads();
  }

  const float w1dq = fmaxf(scal[0] * (1.0f / W_CNT), EPSQ);
  const float w2dq = fmaxf(scal[1] * (1.0f / W_CNT), EPSQ);
  const int n0 = ffBase + wn * 32 + l16;
  #pragma unroll
  for (int m = 0; m < 2; m++) {
    const v8i* g = m ? &acc110 : &acc100;
    const v8i* gn = m ? &acc111 : &acc101;
    const v8i* u = m ? &acc210 : &acc200;
    const v8i* un = m ? &acc211 : &acc201;
    #pragma unroll
    for (int v = 0; v < 8; v++) {
      const int rowt = tokenBase + wmq * 32 + m * 16 + v + half * 8;
      const float xs = xscale[rowt];
      const float d1 = xs * w1dq, d2 = xs * w2dq;
      float g0 = (float)(*g)[v] * d1;
      float m0 = (float)(*u)[v] * d2;
      float h0 = (g0 / (1.0f + __expf(-g0))) * m0;
      float g1 = (float)(*gn)[v] * d1;
      float m1 = (float)(*un)[v] * d2;
      float h1 = (g1 / (1.0f + __expf(-g1))) * m1;
      const long base = (long)rowt * F_DIM;
      h[base + n0] = f32_to_bf16(h0);
      h[base + n0 + 16] = f32_to_bf16(h1);
    }
  }
}

__global__ __launch_bounds__(256)
void gemm3_kernel(const signed char* __restrict__ hq,
                  const signed char* __restrict__ wq3,
                  const float* __restrict__ hscale,
                  const float* __restrict__ scal,
                  float* __restrict__ out) {
  __shared__ int sA[2][128 * 20];
  __shared__ int sB[2][64 * 20];

  const int tid = threadIdx.x;
  const int lane = tid & 31;
  const int wave = tid >> 5;
  const int wmq = wave >> 1;
  const int wn = wave & 1;
  const int half = lane >> 4;
  const int l16 = lane & 15;

  const int tokenBase = blockIdx.y << 7;
  const int eBase = blockIdx.x << 6;

  const int arow = tid >> 1;
  const int acol = (tid & 1) << 5;
  const signed char* ga = hq + (long)(tokenBase + arow) * F_DIM + acol;
  const int sIdxA = arow * 20 + ((tid & 1) << 3);

  const int brow = tid >> 2;
  const int bcol = (tid & 3) << 4;
  const signed char* gb = wq3 + (long)(eBase + brow) * F_DIM + bcol;
  const int sIdxB = brow * 20 + ((tid & 3) << 2);

  const unsigned lA[2] = {lds_off(&sA[0][sIdxA]), lds_off(&sA[1][sIdxA])};
  const unsigned lB[2] = {lds_off(&sB[0][sIdxB]), lds_off(&sB[1][sIdxB])};

  const int aFrag0 = (wmq * 32 + l16) * 20 + half * 4;
  const int aFrag1 = aFrag0 + 16 * 20;
  const int bFrag0 = (wn * 32 + l16) * 20 + half * 4;
  const int bFrag1 = bFrag0 + 16 * 20;

  v8i z = {0, 0, 0, 0, 0, 0, 0, 0};
  v8i acc00 = z, acc01 = z, acc10 = z, acc11 = z;

  ASYNC_B128(lA[0], ga);
  ASYNC_B128(lA[0] + 16u, ga + 16);
  ASYNC_B128(lB[0], gb);

  const int NK = F_DIM / 64;
  for (int i = 0; i < NK; ++i) {
    const int cur = i & 1;
    if (i + 1 < NK) {
      const int nxt = cur ^ 1;
      const int kb = (i + 1) * 64;
      ASYNC_B128(lA[nxt], ga + kb);
      ASYNC_B128(lA[nxt] + 16u, ga + kb + 16);
      ASYNC_B128(lB[nxt], gb + kb);
      WAIT_ASYNC(0x3);
    } else {
      WAIT_ASYNC(0x0);
    }
    __syncthreads();

    const int* As = sA[cur];
    const int* Bs = sB[cur];

    int2 q0 = *(const int2*)&As[aFrag0 + 0];
    int2 q1 = *(const int2*)&As[aFrag0 + 2];
    int2 q2 = *(const int2*)&As[aFrag0 + 8];
    int2 q3 = *(const int2*)&As[aFrag0 + 10];
    v8i a0 = {q0.x, q0.y, q1.x, q1.y, q2.x, q2.y, q3.x, q3.y};
    q0 = *(const int2*)&As[aFrag1 + 0];
    q1 = *(const int2*)&As[aFrag1 + 2];
    q2 = *(const int2*)&As[aFrag1 + 8];
    q3 = *(const int2*)&As[aFrag1 + 10];
    v8i a1 = {q0.x, q0.y, q1.x, q1.y, q2.x, q2.y, q3.x, q3.y};

    int4 p0 = *(const int4*)&Bs[bFrag0];
    int4 p1 = *(const int4*)&Bs[bFrag0 + 8];
    int4 p2 = *(const int4*)&Bs[bFrag1];
    int4 p3 = *(const int4*)&Bs[bFrag1 + 8];
    v8i b0 = {p0.x, p0.y, p0.z, p0.w, p1.x, p1.y, p1.z, p1.w};
    v8i b1 = {p2.x, p2.y, p2.z, p2.w, p3.x, p3.y, p3.z, p3.w};

    acc00 = __builtin_amdgcn_wmma_i32_16x16x64_iu8(true, a0, true, b0, acc00, false, false);
    acc01 = __builtin_amdgcn_wmma_i32_16x16x64_iu8(true, a0, true, b1, acc01, false, false);
    acc10 = __builtin_amdgcn_wmma_i32_16x16x64_iu8(true, a1, true, b0, acc10, false, false);
    acc11 = __builtin_amdgcn_wmma_i32_16x16x64_iu8(true, a1, true, b1, acc11, false, false);

    __syncthreads();
  }

  const float w3dq = fmaxf(scal[2] * (1.0f / W_CNT), EPSQ);
  const int n0 = eBase + wn * 32 + l16;
  #pragma unroll
  for (int m = 0; m < 2; m++) {
    const v8i* c0 = m ? &acc10 : &acc00;
    const v8i* c1 = m ? &acc11 : &acc01;
    #pragma unroll
    for (int v = 0; v < 8; v++) {
      const int rowt = tokenBase + wmq * 32 + m * 16 + v + half * 8;
      const float d = hscale[rowt] * w3dq;
      out[(long)rowt * E_DIM + n0] = (float)(*c0)[v] * d;
      out[(long)rowt * E_DIM + n0 + 16] = (float)(*c1)[v] * d;
    }
  }
}


extern "C" void kernel_launch(void* const* d_in, const int* in_sizes, int n_in,
                              void* d_out, int out_size, void* d_ws, size_t ws_size,
                              hipStream_t stream) {
  (void)in_sizes; (void)n_in; (void)out_size; (void)ws_size;
  const float* x = (const float*)d_in[0];
  const float* w1 = (const float*)d_in[1];
  const float* w2 = (const float*)d_in[2];
  const float* w3 = (const float*)d_in[3];
  float* out = (float*)d_out;

  char* ws = (char*)d_ws;
  float* scal = (float*)ws;
  size_t off = 256;
  signed char* xq = (signed char*)(ws + off);  off += (size_t)T_TOK * E_DIM;
  signed char* wq1 = (signed char*)(ws + off); off += (size_t)F_DIM * E_DIM;
  signed char* wq2 = (signed char*)(ws + off); off += (size_t)F_DIM * E_DIM;
  signed char* wq3 = (signed char*)(ws + off); off += (size_t)E_DIM * F_DIM;
  float* xscale = (float*)(ws + off);          off += (size_t)T_TOK * 4;
  float* hscale = (float*)(ws + off);          off += (size_t)T_TOK * 4;
  signed char* hq = (signed char*)(ws + off);  off += (size_t)T_TOK * F_DIM;
  unsigned short* h = (unsigned short*)(ws + off); off += (size_t)T_TOK * F_DIM * 2;

  const long NW = (long)F_DIM * E_DIM;

  zero_kernel<<<1, 64, 0, stream>>>(scal);
  absum_kernel<<<1024, 256, 0, stream>>>(w1, scal + 0, NW / 4);
  absum_kernel<<<1024, 256, 0, stream>>>(w2, scal + 1, NW / 4);
  absum_kernel<<<1024, 256, 0, stream>>>(w3, scal + 2, NW / 4);
  wquant_kernel<<<(int)(NW / 1024), 256, 0, stream>>>(w1, wq1, scal + 0);
  wquant_kernel<<<(int)(NW / 1024), 256, 0, stream>>>(w2, wq2, scal + 1);
  wquant_kernel<<<(int)(NW / 1024), 256, 0, stream>>>(w3, wq3, scal + 2);
  act_quant_x_kernel<<<T_TOK, 256, 0, stream>>>(x, xq, xscale);
  gemm12_silu_kernel<<<dim3(F_DIM / 64, T_TOK / 128), 256, 0, stream>>>(
      xq, wq1, wq2, xscale, scal, h);
  act_quant_h_kernel<<<T_TOK, 256, 0, stream>>>(h, hq, hscale);
  gemm3_kernel<<<dim3(E_DIM / 64, T_TOK / 128), 256, 0, stream>>>(
      hq, wq3, hscale, scal, out);
}
